// UEMA_Amp_22342419874139
// MI455X (gfx1250) — matrix operand order repaired and checked by reading, not compiled, not run on hardware
//
#include <hip/hip_runtime.h>
#include <math.h>

// ---------------------------------------------------------------------------
// UEMA_Amp fusion block for MI455X (gfx1250), wave32 + WMMA bf16 16x16x32.
//
// All GEMM-like work (1x1 convs, implicit-GEMM 3x3 convs, flash attention)
// runs on v_wmma_f32_16x16x32_bf16 with f32 accumulation. Activations are
// stored NHWC ([pixel][channel], bf16); conv inputs additionally carry a
// 1-pixel zeroed halo ((h+2)x(w+2)) so border taps need NO predication:
// the conv MAC loop is straight-line clause-loads + WMMA (no exec-mask ops,
// no zero-fills, no WAR-hazard NOPs). Fragment layouts used here (A/B: contiguous K):
//   A: lane L -> row M=L&15, K = kh*16+{0..15}, kh=L>>4
//   B: lane L -> col N=L&15, K = kh*16+{0..15}
//   C/D f32: lane L -> col N=L&15, elem v -> row M = v + 8*(L>>4)
// GEMM/conv waves are register-blocked 16(M) x 64(N) and software-pipelined:
// next K-step fragments load while the current step's 4 WMMAs run (partial
// s_wait_loadcnt instead of full drains); K is compile-time so MAC loops
// fully unroll. Attention is flash-style (no 268MB sim matrix): online
// softmax with wave32 shfl_xor half-reductions, P transposed through padded
// LDS, global_prefetch_b8 pulling the next key/value chunk during WMMAs.
// ---------------------------------------------------------------------------

typedef __bf16 bf16;
typedef __attribute__((ext_vector_type(16))) __bf16 v16bf;
typedef __attribute__((ext_vector_type(8)))  __bf16 v8bf;
typedef __attribute__((ext_vector_type(8)))  float  v8f;

#define B_    4
#define UC_   64
#define C_    128
#define H_    128
#define W_    128
#define HX_   64
#define WX_   64
#define N_    4096      // HX_*WX_
#define NPIX_ 16384     // H_*W_
#define QK_SCALE 0.08838834764831845f   // 128^-0.5
#define BN_EPS   1e-5f

static __device__ __forceinline__ int lane_id() { return threadIdx.x & 31; }

static __device__ __forceinline__ v8f zero8f() {
  v8f z;
#pragma unroll
  for (int i = 0; i < 8; ++i) z[i] = 0.0f;
  return z;
}

static __device__ __forceinline__ v16bf pack16(v8bf lo, v8bf hi) {
  v16bf r;
#pragma unroll
  for (int i = 0; i < 8; ++i) { r[i] = lo[i]; r[i + 8] = hi[i]; }
  return r;
}

// A fragment: source row-major [row][k], stride lda (elements).
static __device__ __forceinline__ v16bf load_A(const bf16* __restrict__ base, int lda) {
  int l = lane_id();
  const bf16* p = base + (size_t)(l & 15) * lda + ((l >> 4) << 4);
  return pack16(*(const v8bf*)p, *(const v8bf*)(p + 8));
}

// B fragment: source stored [n][k] (k contiguous), stride ldk (elements).
static __device__ __forceinline__ v16bf load_B_nk(const bf16* __restrict__ base, int ldk) {
  int l = lane_id();
  const bf16* p = base + (size_t)(l & 15) * ldk + ((l >> 4) << 4);
  return pack16(*(const v8bf*)p, *(const v8bf*)(p + 8));
}

static __device__ __forceinline__ v8f wmma_bf(v16bf a, v16bf b, v8f c) {
  return __builtin_amdgcn_wmma_f32_16x16x32_bf16(false, a, false, b, (short)0, c, false, false);
}

static __device__ __forceinline__ float half_max(float v) {
  v = fmaxf(v, __shfl_xor(v, 1, 32));
  v = fmaxf(v, __shfl_xor(v, 2, 32));
  v = fmaxf(v, __shfl_xor(v, 4, 32));
  v = fmaxf(v, __shfl_xor(v, 8, 32));
  return v;
}
static __device__ __forceinline__ float half_sum(float v) {
  v += __shfl_xor(v, 1, 32);
  v += __shfl_xor(v, 2, 32);
  v += __shfl_xor(v, 4, 32);
  v += __shfl_xor(v, 8, 32);
  return v;
}

// ---------------------------------------------------------------------------
// Weight conversion kernels (f32 -> bf16, with layout permutes).
// ---------------------------------------------------------------------------
__global__ void cvt1x1_kernel(const float* __restrict__ w, bf16* __restrict__ o, int n) {
  int i = blockIdx.x * blockDim.x + threadIdx.x;
  if (i < n) o[i] = (bf16)w[i];
}

// [co][ci][3][3] f32 -> [tap][co][ci] bf16  (tap = dy*3+dx), co=ci=128
__global__ void cvt3x3_kernel(const float* __restrict__ w, bf16* __restrict__ o) {
  int i = blockIdx.x * blockDim.x + threadIdx.x;
  if (i < C_ * C_ * 9) {
    int tap = i % 9;
    int ci  = (i / 9) % C_;
    int co  = i / (9 * C_);
    o[(size_t)tap * C_ * C_ + (size_t)co * C_ + ci] = (bf16)w[i];
  }
}

// ---------------------------------------------------------------------------
// Zero the 1-pixel halo of a padded NHWC buffer [b][(h+2)][(w+2)][C_].
// ---------------------------------------------------------------------------
__global__ void halo_kernel(bf16* __restrict__ buf, int h, int w) {
  int b  = blockIdx.y;
  int nh = 2 * (w + 2) + 2 * h;               // halo pixel count
  int i  = blockIdx.x * blockDim.x + threadIdx.x;
  if (i >= nh * C_) return;
  int pix = i / C_;
  int c   = i % C_;
  int row, col;
  if (pix < w + 2)            { row = 0;     col = pix; }
  else if (pix < 2 * (w + 2)) { row = h + 1; col = pix - (w + 2); }
  else {
    int rem = pix - 2 * (w + 2);
    row = 1 + (rem >> 1);
    col = (rem & 1) ? (w + 1) : 0;
  }
  buf[(size_t)b * (h + 2) * (w + 2) * C_ + ((size_t)row * (w + 2) + col) * C_ + c] = (bf16)0.0f;
}

// ---------------------------------------------------------------------------
// 2x2 mean pool x (NCHW f32) -> x_tmp NHWC (f32 + bf16), dense
// ---------------------------------------------------------------------------
__global__ void pool_kernel(const float* __restrict__ x,
                            float* __restrict__ xtF, bf16* __restrict__ xtB) {
  int b   = blockIdx.y;
  int pix = blockIdx.x;         // 0..4095
  int c   = threadIdx.x;        // 0..127
  int y = pix >> 6, xx = pix & 63;
  const float* p = x + (((size_t)b * C_ + c) * H_ + (y << 1)) * W_ + (xx << 1);
  float s = 0.25f * (p[0] + p[1] + p[W_] + p[W_ + 1]);
  size_t o = ((size_t)b * N_ + pix) * C_ + c;
  xtF[o] = s;
  xtB[o] = (bf16)s;
}

// ---------------------------------------------------------------------------
// Bilinear upsample x_uc 32x32 -> 64x64 (align_corners) -> dense NHWC bf16
// ---------------------------------------------------------------------------
__global__ void upuc_kernel(const float* __restrict__ xuc, bf16* __restrict__ xu) {
  int b   = blockIdx.y;
  int pix = blockIdx.x;         // 0..4095
  int c   = threadIdx.x;        // 0..63
  int oy = pix >> 6, ox = pix & 63;
  const float sc = 31.0f / 63.0f;
  float fy = oy * sc, fx = ox * sc;
  int y0 = (int)floorf(fy), x0 = (int)floorf(fx);
  int y1 = min(y0 + 1, 31),  x1 = min(x0 + 1, 31);
  float wy = fy - (float)y0, wx = fx - (float)x0;
  const float* p = xuc + ((size_t)b * UC_ + c) * 1024;
  float v00 = p[y0 * 32 + x0], v01 = p[y0 * 32 + x1];
  float v10 = p[y1 * 32 + x0], v11 = p[y1 * 32 + x1];
  float r0 = v00 * (1.f - wx) + v01 * wx;
  float r1 = v10 * (1.f - wx) + v11 * wx;
  xu[((size_t)b * N_ + pix) * UC_ + c] = (bf16)(r0 * (1.f - wy) + r1 * wy);
}

// ---------------------------------------------------------------------------
// Bilinear upsample fuse (channel-major f32, 64x64) -> padded NHWC bf16
// (interior of a (H_+2)x(W_+2) buffer)
// ---------------------------------------------------------------------------
__global__ void upfuse_kernel(const float* __restrict__ fuse, bf16* __restrict__ fup) {
  int b   = blockIdx.y;
  int pix = blockIdx.x;         // 0..16383
  int c   = threadIdx.x;        // 0..127
  int oy = pix >> 7, ox = pix & 127;
  const float sc = 63.0f / 127.0f;
  float fy = oy * sc, fx = ox * sc;
  int y0 = (int)floorf(fy), x0 = (int)floorf(fx);
  int y1 = min(y0 + 1, 63),  x1 = min(x0 + 1, 63);
  float wy = fy - (float)y0, wx = fx - (float)x0;
  const float* p = fuse + ((size_t)b * C_ + c) * N_;
  float v00 = p[y0 * 64 + x0], v01 = p[y0 * 64 + x1];
  float v10 = p[y1 * 64 + x0], v11 = p[y1 * 64 + x1];
  float r0 = v00 * (1.f - wx) + v01 * wx;
  float r1 = v10 * (1.f - wx) + v11 * wx;
  size_t ob = (size_t)b * (H_ + 2) * (W_ + 2) * C_ +
              ((size_t)(oy + 1) * (W_ + 2) + ox + 1) * C_ + c;
  fup[ob] = (bf16)(r0 * (1.f - wy) + r1 * wy);
}

// ---------------------------------------------------------------------------
// Generic WMMA GEMM: out[b][m][n] = W[m][k] * Bact[b][n][k]  (1x1 convs)
// Register-blocked 16(M) x 64(N); K compile-time; software-pipelined loads.
// mode 0: relu -> bf16 dense NHWC out[n][m]
// mode 1: (resid + relu(acc)) * 0.5 -> bf16 PADDED NHWC (64x64 interior)
// mode 2: relu -> bf16 channel-major out[m][n]   (for V)
// ---------------------------------------------------------------------------
template <int K>
__global__ void gemm_kernel(const bf16* __restrict__ Wm, const bf16* __restrict__ Bact,
                            bf16* __restrict__ outB, const float* __restrict__ resid,
                            int M, int Ntot, int mode) {
  int wid = blockIdx.x * (blockDim.x >> 5) + (threadIdx.x >> 5);
  int tilesM = M >> 4;
  int mt = wid % tilesM;
  int nt = wid / tilesM;
  int b  = blockIdx.z;
  int n0 = nt << 6;                       // 64-wide N block
  const bf16* wp = Wm + (size_t)(mt << 4) * K;
  const bf16* bp = Bact + (size_t)b * Ntot * K + (size_t)n0 * K;

  v8f acc[4];
#pragma unroll
  for (int j = 0; j < 4; ++j) acc[j] = zero8f();

  // software-pipelined MAC loop (fully unrolled, K is compile-time)
  v16bf a_c = load_A(wp, K);
  v16bf b_c[4];
#pragma unroll
  for (int j = 0; j < 4; ++j) b_c[j] = load_B_nk(bp + (size_t)(j << 4) * K, K);

#pragma unroll
  for (int k0 = 0; k0 < K; k0 += 32) {
    v16bf a_n = a_c;
    v16bf b_n[4];
#pragma unroll
    for (int j = 0; j < 4; ++j) b_n[j] = b_c[j];
    if (k0 + 32 < K) {                    // constant-folded per unrolled iter
      a_n = load_A(wp + k0 + 32, K);
#pragma unroll
      for (int j = 0; j < 4; ++j)
        b_n[j] = load_B_nk(bp + (size_t)(j << 4) * K + k0 + 32, K);
    }
#pragma unroll
    for (int j = 0; j < 4; ++j) acc[j] = wmma_bf(a_c, b_c[j], acc[j]);
    a_c = a_n;
#pragma unroll
    for (int j = 0; j < 4; ++j) b_c[j] = b_n[j];
  }

  int l = lane_id();
  int mbase = (mt << 4) + ((l >> 4) << 3);

#pragma unroll
  for (int j = 0; j < 4; ++j) {
    int nc = n0 + (j << 4) + (l & 15);
    if (mode == 2) {                      // channel-major out [M][Ntot]
      size_t ob = (size_t)b * M * Ntot;
#pragma unroll
      for (int v = 0; v < 8; ++v)
        outB[ob + (size_t)(mbase + v) * Ntot + nc] = (bf16)fmaxf(acc[j][v], 0.0f);
    } else if (mode == 1) {               // blend -> padded 64x64 NHWC
      int yy = nc >> 6, xx = nc & 63;
      size_t ob = (size_t)b * (HX_ + 2) * (WX_ + 2) * C_ +
                  ((size_t)(yy + 1) * (WX_ + 2) + xx + 1) * C_ + mbase;
      const float* rp = resid + (size_t)b * Ntot * M + (size_t)nc * M + mbase;
      v8bf o;
#pragma unroll
      for (int v = 0; v < 8; ++v)
        o[v] = (bf16)((rp[v] + fmaxf(acc[j][v], 0.0f)) * 0.5f);
      *(v8bf*)(outB + ob) = o;
    } else {                              // relu -> dense NHWC
      size_t ob = (size_t)b * Ntot * M + (size_t)nc * M + mbase;
      v8bf o;
#pragma unroll
      for (int v = 0; v < 8; ++v)
        o[v] = (bf16)fmaxf(acc[j][v], 0.0f);
      *(v8bf*)(outB + ob) = o;
    }
  }
}

// ---------------------------------------------------------------------------
// Implicit-GEMM 3x3 conv, Cin=Cout=128. Weights [tap][co][ci] bf16.
// Input: PADDED NHWC bf16 [(h+2)][(w+2)][C_] with zeroed halo -> no border
// predication at all; straight-line 9 taps x 4 K-steps x 4 N-blocks WMMA.
// One wave = 16 co x 64 x-pixels of one output row.
// mode 0: relu -> bf16 dense NHWC        (q^T for attention)
// mode 1: relu -> bf16 padded NHWC       (conv chain intermediate)
// mode 2: BN(eval) + relu -> bf16 padded NHWC
// mode 3: + residual (x, NCHW f32) -> f32 NCHW (final output)
// ---------------------------------------------------------------------------
__global__ void conv3x3_kernel(const bf16* __restrict__ Wt, const bf16* __restrict__ in,
                               bf16* __restrict__ outB, float* __restrict__ outF,
                               const float* __restrict__ residNCHW,
                               const float* __restrict__ gamma, const float* __restrict__ beta,
                               const float* __restrict__ mean, const float* __restrict__ var,
                               int h, int w, int mode) {
  const int Cout = C_;
  int wid = blockIdx.x * (blockDim.x >> 5) + (threadIdx.x >> 5);
  int tilesX = w >> 6;               // 64-wide pixel blocks: 1 (hx) or 2 (h)
  int mt = wid & 7;                  // Cout/16 == 8
  int xt = (wid >> 3) % tilesX;
  int y  = wid / (8 * tilesX);       // output row
  int b  = blockIdx.z;
  int x0 = xt << 6;

  int l  = lane_id();
  int ln = l & 15;
  int kh = l >> 4;
  int pw = w + 2;                    // padded row stride (pixels)
  size_t inP = (size_t)b * (h + 2) * pw * C_;

  v8f acc[4];
#pragma unroll
  for (int j = 0; j < 4; ++j) acc[j] = zero8f();

#pragma unroll
  for (int dy = 0; dy < 3; ++dy) {
    // padded row index y+dy in [0, h+1]; halo rows are zero -> no check
    const bf16* rowp = in + inP + (size_t)(y + dy) * pw * C_ + (kh << 4);
#pragma unroll
    for (int dx = 0; dx < 3; ++dx) {
      const bf16* wbase = Wt + (size_t)(dy * 3 + dx) * Cout * C_ + (size_t)(mt << 4) * C_;
      // padded col x0+dx+ln+j*16 in [0, w+1]; halo cols are zero -> no check
      const bf16* pl[4];
#pragma unroll
      for (int j = 0; j < 4; ++j)
        pl[j] = rowp + (size_t)(x0 + dx + ln + (j << 4)) * C_;

      // software-pipelined over the 4 K-steps of this tap
      v16bf a_c = load_A(wbase, C_);
      v16bf b_c[4];
#pragma unroll
      for (int j = 0; j < 4; ++j)
        b_c[j] = pack16(*(const v8bf*)pl[j], *(const v8bf*)(pl[j] + 8));
#pragma unroll
      for (int k0 = 0; k0 < C_; k0 += 32) {
        v16bf a_n = a_c;
        v16bf b_n[4];
#pragma unroll
        for (int j = 0; j < 4; ++j) b_n[j] = b_c[j];
        if (k0 + 32 < C_) {
          a_n = load_A(wbase + k0 + 32, C_);
#pragma unroll
          for (int j = 0; j < 4; ++j)
            b_n[j] = pack16(*(const v8bf*)(pl[j] + k0 + 32),
                            *(const v8bf*)(pl[j] + k0 + 40));
        }
#pragma unroll
        for (int j = 0; j < 4; ++j) acc[j] = wmma_bf(a_c, b_c[j], acc[j]);
        a_c = a_n;
#pragma unroll
        for (int j = 0; j < 4; ++j) b_c[j] = b_n[j];
      }
    }
  }

  int mbase = (mt << 4) + (kh << 3);                 // channel base
#pragma unroll
  for (int j = 0; j < 4; ++j) {
    int xc = x0 + (j << 4) + ln;                     // output col
    if (mode == 0) {                                 // dense NHWC
      size_t ob = (size_t)b * h * w * Cout + ((size_t)y * w + xc) * Cout + mbase;
      v8bf o;
#pragma unroll
      for (int v = 0; v < 8; ++v) o[v] = (bf16)fmaxf(acc[j][v], 0.0f);
      *(v8bf*)(outB + ob) = o;
    } else if (mode == 1) {                          // padded NHWC
      size_t ob = (size_t)b * (h + 2) * pw * Cout +
                  ((size_t)(y + 1) * pw + xc + 1) * Cout + mbase;
      v8bf o;
#pragma unroll
      for (int v = 0; v < 8; ++v) o[v] = (bf16)fmaxf(acc[j][v], 0.0f);
      *(v8bf*)(outB + ob) = o;
    } else if (mode == 2) {                          // BN + relu, padded NHWC
      size_t ob = (size_t)b * (h + 2) * pw * Cout +
                  ((size_t)(y + 1) * pw + xc + 1) * Cout + mbase;
      v8bf o;
#pragma unroll
      for (int v = 0; v < 8; ++v) {
        int ch = mbase + v;
        float inv = gamma[ch] * rsqrtf(var[ch] + BN_EPS);
        float valb = acc[j][v] * inv + (beta[ch] - mean[ch] * inv);
        o[v] = (bf16)fmaxf(valb, 0.0f);
      }
      *(v8bf*)(outB + ob) = o;
    } else {                                         // mode 3: residual, NCHW f32
      int nc = y * w + xc;
      size_t ob = (size_t)b * Cout * h * w;
#pragma unroll
      for (int v = 0; v < 8; ++v) {
        int ch = mbase + v;
        size_t idx = ob + (size_t)ch * h * w + nc;
        outF[idx] = residNCHW[idx] + acc[j][v];
      }
    }
  }
}

// ---------------------------------------------------------------------------
// Flash attention: q^T [n][c] bf16, k^T [m][c] bf16, v [c][m] bf16 ->
// fuse [c][n] f32. One wave per 16 query rows, keys chunked by 32.
// ---------------------------------------------------------------------------
__global__ void __launch_bounds__(32)
attn_kernel(const bf16* __restrict__ qT, const bf16* __restrict__ kT,
            const bf16* __restrict__ vC, float* __restrict__ fuse) {
  __shared__ __align__(16) bf16 sP[16][48];          // P tile [nq][m], padded rows
  __shared__ float sF[16];                           // per-column rescale factors
  __shared__ float sS[16];                           // per-column softmax sums

  int b   = blockIdx.y;
  int nq0 = blockIdx.x << 4;
  int l   = lane_id();
  int ln  = l & 15;
  int kh  = l >> 4;

  // preload A = q^T tile (16 rows x 128 c) as 4 fragments
  const bf16* qb = qT + (size_t)b * N_ * C_ + (size_t)nq0 * C_;
  v16bf aq[4];
#pragma unroll
  for (int kc = 0; kc < 4; ++kc) aq[kc] = load_A(qb + kc * 32, C_);

  v8f O[8];
#pragma unroll
  for (int t = 0; t < 8; ++t) O[t] = zero8f();
  float rmax[8], rsum[8];
#pragma unroll
  for (int v = 0; v < 8; ++v) { rmax[v] = -1e30f; rsum[v] = 0.0f; }

  const bf16* kb = kT + (size_t)b * N_ * C_;
  const bf16* vb = vC + (size_t)b * C_ * N_;

  for (int m0 = 0; m0 < N_; m0 += 32) {
    // prefetch next key chunk (one line per lane) while this chunk computes
    int mnext = (m0 + 32 < N_) ? (m0 + 32) : m0;
    __builtin_prefetch(kb + (size_t)(mnext + l) * C_, 0, 1);

    // ---- S = q^T k, software-pipelined over the 4 c-steps ----
    v8f s0 = zero8f(), s1 = zero8f();
    v16bf bk0 = load_B_nk(kb + (size_t)m0 * C_, C_);
    v16bf bk1 = load_B_nk(kb + (size_t)(m0 + 16) * C_, C_);
#pragma unroll
    for (int kc = 0; kc < 4; ++kc) {
      v16bf nb0 = bk0, nb1 = bk1;
      if (kc < 3) {
        nb0 = load_B_nk(kb + (size_t)m0 * C_ + (kc + 1) * 32, C_);
        nb1 = load_B_nk(kb + (size_t)(m0 + 16) * C_ + (kc + 1) * 32, C_);
      }
      s0 = wmma_bf(aq[kc], bk0, s0);
      s1 = wmma_bf(aq[kc], bk1, s1);
      bk0 = nb0;
      bk1 = nb1;
    }

    // ---- online softmax row stats (rows live across 16 lanes of a half) ----
    float p0[8], p1[8], fac[8];
#pragma unroll
    for (int v = 0; v < 8; ++v) {
      float a0 = s0[v] * QK_SCALE, a1 = s1[v] * QK_SCALE;
      float cm = half_max(fmaxf(a0, a1));
      float nm = fmaxf(rmax[v], cm);
      fac[v] = __expf(rmax[v] - nm);
      p0[v]  = __expf(a0 - nm);
      p1[v]  = __expf(a1 - nm);
      float ps = half_sum(p0[v] + p1[v]);
      rsum[v] = rsum[v] * fac[v] + ps;
      rmax[v] = nm;
    }

    // route per-row factor to per-column lanes via LDS
    if (l == 0)  {
#pragma unroll
      for (int v = 0; v < 8; ++v) sF[v] = fac[v];
    }
    if (l == 16) {
#pragma unroll
      for (int v = 0; v < 8; ++v) sF[8 + v] = fac[v];
    }
    // stage P (bf16) to LDS: sP[nq_row][m_local]
#pragma unroll
    for (int v = 0; v < 8; ++v) {
      int row = v + (kh << 3);
      sP[row][ln]      = (bf16)p0[v];
      sP[row][ln + 16] = (bf16)p1[v];
    }
    __syncthreads();   // single-wave WG: S_NOP + compiler fence

    float facN = sF[ln];
    v16bf pb;
    {
      const bf16* pp = &sP[ln][kh << 4];
      pb = pack16(*(const v8bf*)pp, *(const v8bf*)(pp + 8));
    }

    // prefetch next value chunk rows
    __builtin_prefetch(vb + (size_t)((l & 7) << 4) * N_ + mnext + ((l >> 3) << 3), 0, 1);

    // ---- O = O*fac + V P^T, pipelined over the 8 channel tiles ----
    v16bf av_c = load_A(vb + m0, N_);
#pragma unroll
    for (int ct = 0; ct < 8; ++ct) {
      v16bf av_n = av_c;
      if (ct < 7) av_n = load_A(vb + (size_t)((ct + 1) << 4) * N_ + m0, N_);
#pragma unroll
      for (int v = 0; v < 8; ++v) O[ct][v] *= facN;   // VALU co-executes w/ WMMA
      O[ct] = wmma_bf(av_c, pb, O[ct]);
      av_c = av_n;
    }
    __syncthreads();
  }

  // normalize and store fuse [c][n] f32
  if (l == 0)  {
#pragma unroll
    for (int v = 0; v < 8; ++v) sS[v] = rsum[v];
  }
  if (l == 16) {
#pragma unroll
    for (int v = 0; v < 8; ++v) sS[8 + v] = rsum[v];
  }
  __syncthreads();
  float invS = 1.0f / sS[ln];
  size_t fb = (size_t)b * C_ * N_;
#pragma unroll
  for (int ct = 0; ct < 8; ++ct)
#pragma unroll
    for (int v = 0; v < 8; ++v) {
      int ch = (ct << 4) + (kh << 3) + v;
      fuse[fb + (size_t)ch * N_ + nq0 + ln] = O[ct][v] * invS;
    }
}

// ---------------------------------------------------------------------------
// Host-side orchestration
// ---------------------------------------------------------------------------
extern "C" void kernel_launch(void* const* d_in, const int* in_sizes, int n_in,
                              void* d_out, int out_size, void* d_ws, size_t ws_size,
                              hipStream_t stream) {
  (void)in_sizes; (void)n_in; (void)out_size; (void)ws_size;
  const float* x_uc  = (const float*)d_in[0];
  const float* x     = (const float*)d_in[1];
  const float* W_pre = (const float*)d_in[2];
  const float* Wq1   = (const float*)d_in[3];
  const float* Wq2   = (const float*)d_in[4];
  const float* Wk1   = (const float*)d_in[5];
  const float* Wk2   = (const float*)d_in[6];
  const float* Wv1   = (const float*)d_in[7];
  const float* Wv2   = (const float*)d_in[8];
  const float* Wc1   = (const float*)d_in[9];
  const float* bn_g  = (const float*)d_in[10];
  const float* bn_b  = (const float*)d_in[11];
  const float* bn_m  = (const float*)d_in[12];
  const float* bn_v  = (const float*)d_in[13];
  const float* Wc2   = (const float*)d_in[14];
  float* out = (float*)d_out;

  const size_t padS  = (size_t)(HX_ + 2) * (WX_ + 2) * C_;   // small padded plane
  const size_t padL  = (size_t)(H_ + 2) * (W_ + 2) * C_;     // large padded plane

  char* wsp = (char*)d_ws;
  auto carve = [&](size_t bytes) -> char* {
    char* p = wsp;
    wsp += (bytes + 255) & ~(size_t)255;
    return p;
  };
  bf16* wpre = (bf16*)carve((size_t)C_ * UC_ * 2);
  bf16* wq1b = (bf16*)carve((size_t)9 * C_ * C_ * 2);
  bf16* wq2b = (bf16*)carve((size_t)9 * C_ * C_ * 2);
  bf16* wc1b = (bf16*)carve((size_t)9 * C_ * C_ * 2);
  bf16* wc2b = (bf16*)carve((size_t)9 * C_ * C_ * 2);
  bf16* wk1b = (bf16*)carve((size_t)C_ * C_ * 2);
  bf16* wk2b = (bf16*)carve((size_t)C_ * C_ * 2);
  bf16* wv1b = (bf16*)carve((size_t)C_ * C_ * 2);
  bf16* wv2b = (bf16*)carve((size_t)C_ * C_ * 2);
  float* xtF  = (float*)carve((size_t)B_ * N_ * C_ * 4);
  bf16*  xtB  = (bf16*) carve((size_t)B_ * N_ * C_ * 2);
  bf16*  xu   = (bf16*) carve((size_t)B_ * N_ * UC_ * 2);
  bf16*  qinP = (bf16*) carve((size_t)B_ * padS * 2);   // padded conv input
  bf16*  tCvP = (bf16*) carve((size_t)B_ * padS * 2);   // padded conv intermediate
  bf16*  tG   = (bf16*) carve((size_t)B_ * N_ * C_ * 2);// dense gemm intermediate
  bf16*  qTb  = (bf16*) carve((size_t)B_ * N_ * C_ * 2);
  bf16*  kTb  = (bf16*) carve((size_t)B_ * N_ * C_ * 2);
  bf16*  vCb  = (bf16*) carve((size_t)B_ * C_ * N_ * 2);
  float* fuse = (float*)carve((size_t)B_ * C_ * N_ * 4);
  bf16*  fupP = (bf16*) carve((size_t)B_ * padL * 2);   // padded conv input
  bf16*  y1P  = (bf16*) carve((size_t)B_ * padL * 2);   // padded conv intermediate

  // weight conversion
  cvt1x1_kernel<<<(C_ * UC_ + 255) / 256, 256, 0, stream>>>(W_pre, wpre, C_ * UC_);
  cvt1x1_kernel<<<(C_ * C_ + 255) / 256, 256, 0, stream>>>(Wk1, wk1b, C_ * C_);
  cvt1x1_kernel<<<(C_ * C_ + 255) / 256, 256, 0, stream>>>(Wk2, wk2b, C_ * C_);
  cvt1x1_kernel<<<(C_ * C_ + 255) / 256, 256, 0, stream>>>(Wv1, wv1b, C_ * C_);
  cvt1x1_kernel<<<(C_ * C_ + 255) / 256, 256, 0, stream>>>(Wv2, wv2b, C_ * C_);
  int n9 = 9 * C_ * C_;
  cvt3x3_kernel<<<(n9 + 255) / 256, 256, 0, stream>>>(Wq1, wq1b);
  cvt3x3_kernel<<<(n9 + 255) / 256, 256, 0, stream>>>(Wq2, wq2b);
  cvt3x3_kernel<<<(n9 + 255) / 256, 256, 0, stream>>>(Wc1, wc1b);
  cvt3x3_kernel<<<(n9 + 255) / 256, 256, 0, stream>>>(Wc2, wc2b);

  // zero halos of padded buffers (interior overwritten by producers)
  {
    int nhS = (2 * (WX_ + 2) + 2 * HX_) * C_;
    int nhL = (2 * (W_ + 2) + 2 * H_) * C_;
    halo_kernel<<<dim3((nhS + 255) / 256, B_), 256, 0, stream>>>(qinP, HX_, WX_);
    halo_kernel<<<dim3((nhS + 255) / 256, B_), 256, 0, stream>>>(tCvP, HX_, WX_);
    halo_kernel<<<dim3((nhL + 255) / 256, B_), 256, 0, stream>>>(fupP, H_, W_);
    halo_kernel<<<dim3((nhL + 255) / 256, B_), 256, 0, stream>>>(y1P, H_, W_);
  }

  // pooling + uc upsample
  pool_kernel<<<dim3(N_, B_), C_, 0, stream>>>(x, xtF, xtB);
  upuc_kernel<<<dim3(N_, B_), UC_, 0, stream>>>(x_uc, xu);

  // q_in = (x_tmp + relu(W_pre @ x_uc_up)) / 2 -> padded NHWC
  // waves/batch = (M/16)*(N/64) = 8*64 = 512 -> 128 blocks of 4 waves
  gemm_kernel<UC_><<<dim3(128, 1, B_), 128, 0, stream>>>(wpre, xu, qinP, xtF, C_, N_, 1);

  // uc_q = relu(conv3(relu(conv3(q_in, Wq1)), Wq2))  -> q^T layout (dense NHWC)
  // waves/batch = 8 * (64/64) * 64 = 512 -> 128 blocks
  conv3x3_kernel<<<dim3(128, 1, B_), 128, 0, stream>>>(wq1b, qinP, tCvP, nullptr, nullptr,
                                                       nullptr, nullptr, nullptr, nullptr,
                                                       HX_, WX_, 1);
  conv3x3_kernel<<<dim3(128, 1, B_), 128, 0, stream>>>(wq2b, tCvP, qTb, nullptr, nullptr,
                                                       nullptr, nullptr, nullptr, nullptr,
                                                       HX_, WX_, 0);

  // k = relu(Wk2 relu(Wk1 x_tmp))  -> k^T (dense NHWC); v -> channel-major
  gemm_kernel<C_><<<dim3(128, 1, B_), 128, 0, stream>>>(wk1b, xtB, tG, nullptr, C_, N_, 0);
  gemm_kernel<C_><<<dim3(128, 1, B_), 128, 0, stream>>>(wk2b, tG, kTb, nullptr, C_, N_, 0);
  gemm_kernel<C_><<<dim3(128, 1, B_), 128, 0, stream>>>(wv1b, xtB, tG, nullptr, C_, N_, 0);
  gemm_kernel<C_><<<dim3(128, 1, B_), 128, 0, stream>>>(wv2b, tG, vCb, nullptr, C_, N_, 2);

  // flash attention -> fuse [c][n] f32
  attn_kernel<<<dim3(N_ / 16, B_), 32, 0, stream>>>(qTb, kTb, vCb, fuse);

  // upsample back to 128x128 -> padded NHWC
  upfuse_kernel<<<dim3(NPIX_, B_), C_, 0, stream>>>(fuse, fupP);

  // conv3x3 + BN + relu (padded out), then conv3x3 + residual -> d_out (NCHW f32)
  // waves/batch = 8 * (128/64) * 128 = 2048 -> 512 blocks
  conv3x3_kernel<<<dim3(512, 1, B_), 128, 0, stream>>>(wc1b, fupP, y1P, nullptr, nullptr,
                                                       bn_g, bn_b, bn_m, bn_v,
                                                       H_, W_, 2);
  conv3x3_kernel<<<dim3(512, 1, B_), 128, 0, stream>>>(wc2b, y1P, nullptr, out, x,
                                                       nullptr, nullptr, nullptr, nullptr,
                                                       H_, W_, 3);
}
